// SensorTokenizer_8787503088041
// MI455X (gfx1250) — matrix operand order repaired and checked by reading, not compiled, not run on hardware
//
#include <hip/hip_runtime.h>
#include <hip/hip_bf16.h>

// Problem constants (from setup_inputs): B=4, C=2, T=8, H=128, W=128 -> N=131072
// D=512, P=32 -> F = C+3P = 98 (K-padded to 128), M_max = 65536.
#define NB    4
#define NVOX  131072          // T*H*W
#define MM    65536           // M_max
#define DD    512
#define KPAD  128
#define FREAL 98

typedef __attribute__((ext_vector_type(8)))  __bf16 v8bf;
typedef __attribute__((ext_vector_type(16))) __bf16 v16bf;
typedef __attribute__((ext_vector_type(8)))  float  v8f;

#define SHUF16(lo, hi) __builtin_shufflevector((lo), (hi), 0,1,2,3,4,5,6,7,8,9,10,11,12,13,14,15)

// ---------------------------------------------------------------------------
// Kernel 0: convert proj_w (512 x 98 f32, row-major) -> bf16 512 x 128 (K-padded)
// ---------------------------------------------------------------------------
__global__ void st_wconv(const float* __restrict__ w, __bf16* __restrict__ wbuf) {
    int t = blockIdx.x * blockDim.x + threadIdx.x;   // 0 .. 512*128-1
    int d = t >> 7;
    int f = t & 127;
    float v = (f < FREAL) ? w[d * FREAL + f] : 0.0f;
    wbuf[t] = (__bf16)v;
}

// ---------------------------------------------------------------------------
// Kernel 1: ordered stream compaction per batch (one 1024-thread block / batch).
// Writes compacted voxel indices (or -1) and the pad-mask slice of d_out.
// ---------------------------------------------------------------------------
__global__ __launch_bounds__(1024) void st_compact(const int* __restrict__ mask,
                                                   int* __restrict__ idxbuf,
                                                   float* __restrict__ pad_mask) {
    int b = blockIdx.x;
    const int* mb = mask + (size_t)b * NVOX;
    __shared__ int wsum[32];
    __shared__ int woff[32];
    __shared__ int ctot;

    int tid = threadIdx.x;
    int lane = tid & 31;
    int wid = tid >> 5;
    int running = 0;

    for (int base = 0; base < NVOX; base += 1024) {
        int i = base + tid;
        bool p = (mb[i] != 0);                       // NVOX % 1024 == 0, no tail
        unsigned bal = __builtin_amdgcn_ballot_w32(p);
        int prefix = __popc(bal & ((1u << lane) - 1u));
        int wt = __popc(bal);
        if (lane == 0) wsum[wid] = wt;
        __syncthreads();
        if (tid == 0) {
            int s = 0;
            for (int w2 = 0; w2 < 32; ++w2) { woff[w2] = s; s += wsum[w2]; }
            ctot = s;
        }
        __syncthreads();
        if (p) {
            int pos = running + woff[wid] + prefix;
            if (pos < MM) idxbuf[(size_t)b * MM + pos] = i;
        }
        running += ctot;
        __syncthreads();                              // protect wsum/ctot reuse
    }

    int count = (running < MM) ? running : MM;
    for (int m = tid; m < MM; m += 1024) {
        bool padv = (m >= count);
        if (padv) idxbuf[(size_t)b * MM + m] = -1;
        pad_mask[(size_t)b * MM + m] = padv ? 1.0f : 0.0f;
    }
}

// ---------------------------------------------------------------------------
// Kernel 2: fused feature-build + bf16 WMMA GEMM.
// Block = 256 threads (8 waves). Each wave owns a 32-row slab (two 16-row
// M-tiles register-blocked) and sweeps all 32 N-tiles of D=512 with
// v_wmma_f32_16x16x32_bf16, K = 128 (4 K-blocks). Each B-fragment LDS load
// is amortized over 2 WMMAs. Full weight matrix (bf16 512x128 = 128 KB)
// staged in LDS per workgroup; feat tiles 8 KB/wave. LDS total = 192 KB.
// ---------------------------------------------------------------------------
__global__ __launch_bounds__(256) void st_gemm(const float* __restrict__ y,
                                               const float* __restrict__ pbias,
                                               const __bf16* __restrict__ wbuf,
                                               const int* __restrict__ idxbuf,
                                               float* __restrict__ out) {
    __shared__ __align__(16) __bf16 sW[DD * KPAD];        // 128 KB
    __shared__ __align__(16) __bf16 sF[8][32 * KPAD];     // 8 waves * 8 KB

    int tid = threadIdx.x;
    int lane = tid & 31;
    int wid = tid >> 5;

    // --- cooperative load of full weight matrix into LDS (16B vectors) ---
    {
        const v8bf* src = (const v8bf*)wbuf;
        v8bf* dst = (v8bf*)sW;
        #pragma unroll 4
        for (int t = tid; t < (DD * KPAD) / 8; t += 256) dst[t] = src[t];
    }

    int row0 = (blockIdx.x * 8 + wid) * 32;               // global row in [0, B*M)
    int b = row0 >> 16;                                   // / MM
    int m0 = row0 & (MM - 1);

    // --- build 32 x 128 bf16 feature tile for this wave ---
    for (int r = 0; r < 32; ++r) {
        int idx = idxbuf[(size_t)b * MM + (m0 + r)];
        int tt = 0, hh = 0, ww = 0;
        float v0 = 0.0f, v1 = 0.0f;
        if (idx >= 0) {
            tt = idx >> 14;                               // / (H*W)
            hh = (idx >> 7) & 127;
            ww = idx & 127;
            const float* yb = y + (size_t)b * 2 * NVOX;
            v0 = yb[idx];
            v1 = yb[NVOX + idx];
        }
        #pragma unroll
        for (int j = 0; j < 4; ++j) {
            int k = j * 32 + lane;
            float val = 0.0f;
            if (idx >= 0) {
                if (k == 0) val = v0;
                else if (k == 1) val = v1;
                else if (k < FREAL) {
                    int q = k - 2;
                    int seg = q >> 5;                     // 0:t 1:h 2:w
                    int p = q & 31;
                    int pos = (seg == 0) ? tt : (seg == 1) ? hh : ww;
                    // omega_p = 10000^(-p/16) = exp(-p * ln(10000)/16)
                    float omega = __expf(-(float)(p & 15) * 0.57564627324851148f);
                    float ang = (float)pos * omega;
                    val = (p & 16) ? __cosf(ang) : __sinf(ang);
                }
            }
            sF[wid][r * KPAD + k] = (__bf16)val;
        }
    }
    __syncthreads();

    // --- A fragments: 2 M-tiles x 4 K-blocks, kept in registers (ISA 7.12.2) ---
    // lane (g = lane>>4, r = lane&15): VGPR halves hold K = k0 + {0..7|16..23} + 8g
    int rr = lane & 15;
    int g = lane >> 4;
    v16bf a[2][4];
    #pragma unroll
    for (int mt = 0; mt < 2; ++mt) {
        #pragma unroll
        for (int kb = 0; kb < 4; ++kb) {
            const __bf16* fp = &sF[wid][(mt * 16 + rr) * KPAD + kb * 32 + 8 * g];
            v8bf lo = *(const v8bf*)fp;                   // K = k0+8g .. +7
            v8bf hi = *(const v8bf*)(fp + 16);            // K = k0+16+8g .. +7
            a[mt][kb] = SHUF16(lo, hi);
        }
    }

    // --- sweep 32 N-tiles; each B fragment feeds both M-tiles ---
    #pragma unroll 2
    for (int nt = 0; nt < 32; ++nt) {
        int d0 = nt * 16;
        v8f c0 = {};
        v8f c1 = {};
        #pragma unroll
        for (int kb = 0; kb < 4; ++kb) {
            // B fragment 32x16: lane holds column n=rr, K = k0 + {0..7|16..23} + 8g
            const __bf16* wp = &sW[(d0 + rr) * KPAD + kb * 32 + 8 * g];
            v8bf lo = *(const v8bf*)wp;
            v8bf hi = *(const v8bf*)(wp + 16);
            v16bf bb = SHUF16(lo, hi);
            c0 = __builtin_amdgcn_wmma_f32_16x16x32_bf16(
                     false, a[0][kb], false, bb, (short)0, c0, false, false);
            c1 = __builtin_amdgcn_wmma_f32_16x16x32_bf16(
                     false, a[1][kb], false, bb, (short)0, c1, false, false);
        }
        float bias = pbias[d0 + rr];
        float* orow = out + (size_t)row0 * DD + d0 + rr;
        #pragma unroll
        for (int i = 0; i < 8; ++i) {
            // C/D layout: VGPR i -> M = i + 8g (+16 for second M-tile), N = rr
            orow[(size_t)(i + 8 * g) * DD]      = c0[i] + bias;
            orow[(size_t)(16 + i + 8 * g) * DD] = c1[i] + bias;
        }
    }
}

// ---------------------------------------------------------------------------
extern "C" void kernel_launch(void* const* d_in, const int* in_sizes, int n_in,
                              void* d_out, int out_size, void* d_ws, size_t ws_size,
                              hipStream_t stream) {
    const float* y_obs  = (const float*)d_in[0];   // (4,2,8,128,128) f32
    const int*   mask1  = (const int*)d_in[1];     // (4,1,8,128,128) i32
    const float* proj_w = (const float*)d_in[2];   // (512,98) f32
    const float* proj_b = (const float*)d_in[3];   // (512,) f32
    // d_in[4] = M_max scalar (65536, hardcoded)

    float* out = (float*)d_out;                            // tokens then pad_mask
    float* pad = out + (size_t)NB * MM * DD;

    __bf16* wbuf  = (__bf16*)d_ws;                         // 512*128*2 = 128 KB
    int*    idxbuf = (int*)((char*)d_ws + (size_t)DD * KPAD * 2);  // 4*65536*4 = 1 MB

    st_wconv<<<(DD * KPAD) / 256, 256, 0, stream>>>(proj_w, wbuf);
    st_compact<<<NB, 1024, 0, stream>>>(mask1, idxbuf, pad);

    // total rows = B*M = 262144; 8 waves/block * 32 rows/wave -> 1024 blocks
    st_gemm<<<1024, 256, 0, stream>>>(y_obs, proj_b, wbuf, idxbuf, out);
}
